// LSTM_base_81999515615464
// MI455X (gfx1250) — matrix operand order repaired and checked by reading, not compiled, not run on hardware
//
#include <hip/hip_runtime.h>
#include <hip/hip_bf16.h>
#include <cstdint>

// ---------------------------------------------------------------------------
// Fused persistent LSTM scan for MI455X (gfx1250, wave32, WMMA + async LDS DMA)
//   iou_t = [x_t | h_{t-1}] * Wcomb_iou^T + b ;  f_t similarly -> one GEMM/step
//   Wcomb (2048 x 1024, bf16) sharded across 64 WGs, resident in LDS all T steps.
//   x pre-converted to bf16 once (parallel); per-step staging is pure async DMA
//   (global_load_async_to_lds_b128) double-buffered against the WMMA pipeline.
// ---------------------------------------------------------------------------

#define T_DIM   512
#define B_DIM   64
#define IN_DIM  512
#define M_DIM   512
#define K_DIM   1024          // IN_DIM + M_DIM
#define NWG     64            // workgroups in persistent grid
#define MS      8             // hidden units per WG (M_DIM / NWG)
#define RROWS   32            // 4 gates * MS weight rows per WG
#define NTHREAD 256           // 8 waves of 32
#define CHUNK   256           // K staged per chunk
#define NCHUNK  (K_DIM / CHUNK)
#define KSTEP   32            // WMMA K per instruction

typedef __attribute__((ext_vector_type(16))) __bf16 v16bf;
typedef __attribute__((ext_vector_type(8)))  __bf16 v8bf;
typedef __attribute__((ext_vector_type(8)))  float  v8f;

// ---------------------------------------------------------------------------
// Init: zero barrier counter, h0 -> bf16 hbuf, combine biases.
// ---------------------------------------------------------------------------
__global__ void lstm_init_kernel(const float* __restrict__ h0,
                                 const float* __restrict__ b_ioux,
                                 const float* __restrict__ b_iouh,
                                 const float* __restrict__ b_fx,
                                 const float* __restrict__ b_fh,
                                 __bf16* __restrict__ hbuf,
                                 float* __restrict__ biasc,
                                 unsigned int* __restrict__ ctr) {
  int idx = blockIdx.x * blockDim.x + threadIdx.x;
  if (idx == 0) *ctr = 0u;
  if (idx < B_DIM * M_DIM) hbuf[idx] = (__bf16)h0[idx];
  if (idx < 3 * M_DIM) {
    biasc[idx] = b_ioux[idx] + b_iouh[idx];
  } else if (idx < 4 * M_DIM) {
    int m = idx - 3 * M_DIM;
    biasc[idx] = b_fx[m] + b_fh[m];
  }
}

// ---------------------------------------------------------------------------
// Convert the whole input sequence to bf16 once (parallel over T*B*IN).
// ---------------------------------------------------------------------------
__global__ void lstm_convert_x_kernel(const float* __restrict__ x,
                                      __bf16* __restrict__ xbf) {
  int idx = blockIdx.x * blockDim.x + threadIdx.x;   // one float4 per thread
  size_t base = (size_t)idx * 4;
  if (base + 3 >= (size_t)T_DIM * B_DIM * IN_DIM) return;
  const float4 v = *(const float4*)&x[base];
  xbf[base + 0] = (__bf16)v.x;
  xbf[base + 1] = (__bf16)v.y;
  xbf[base + 2] = (__bf16)v.z;
  xbf[base + 3] = (__bf16)v.w;
}

// ---------------------------------------------------------------------------
// Pack combined weights:
//   rows 0..1535      : [ W_ioux[g][0:512] | W_iouh[g][0:512] ]
//   rows 1536..2047(m): [ W_fx[m][0:512]   | W_fh[m][0:512]   ]
// ---------------------------------------------------------------------------
__global__ void lstm_pack_weights_kernel(const float* __restrict__ W_ioux,
                                         const float* __restrict__ W_iouh,
                                         const float* __restrict__ W_fx,
                                         const float* __restrict__ W_fh,
                                         __bf16* __restrict__ Wcomb) {
  int idx = blockIdx.x * blockDim.x + threadIdx.x;
  if (idx >= 4 * M_DIM * K_DIM) return;
  int r = idx >> 10;          // / K_DIM
  int k = idx & (K_DIM - 1);
  float v;
  if (r < 3 * M_DIM) {
    v = (k < IN_DIM) ? W_ioux[r * IN_DIM + k] : W_iouh[r * M_DIM + (k - IN_DIM)];
  } else {
    int m = r - 3 * M_DIM;
    v = (k < IN_DIM) ? W_fx[m * IN_DIM + k] : W_fh[m * M_DIM + (k - IN_DIM)];
  }
  Wcomb[idx] = (__bf16)v;
}

// ---------------------------------------------------------------------------
// Persistent scan kernel. Grid = NWG blocks x NTHREAD threads (8 waves).
// WG wg owns hidden slice m in [wg*MS, wg*MS+MS).
// ---------------------------------------------------------------------------
__global__ __launch_bounds__(NTHREAD, 1)
void lstm_scan_kernel(const __bf16* __restrict__ xbf,   // [T,B,IN] bf16
                      const float* __restrict__ c0,     // [B,M]
                      const __bf16* __restrict__ Wcomb, // [4M, K]
                      const float* __restrict__ biasc,  // [4M]
                      __bf16* __restrict__ hbuf,        // [B,M] bf16
                      unsigned int* __restrict__ ctr,   // barrier counter
                      float* __restrict__ out) {        // [T*B*M + 2*B*M]
  __shared__ __attribute__((aligned(32))) __bf16 lW[RROWS][K_DIM];      // 64 KB
  __shared__ __attribute__((aligned(32))) __bf16 lXH[2][B_DIM][CHUNK];  // 64 KB
  __shared__ __attribute__((aligned(16))) __bf16 lHs[B_DIM][MS];        // 1 KB
  __shared__ float lG[B_DIM][RROWS];                                    // 8 KB
  __shared__ float lC[B_DIM][MS];                                       // 2 KB

  const int wg   = blockIdx.x;
  const int tid  = threadIdx.x;
  const int lane = tid & 31;
  const int wave = tid >> 5;       // 0..7
  const int m0   = wg * MS;

  // ---- Load this WG's 32 weight rows into LDS (bf16), row order [i,o,u,f] ----
  for (int idx = tid; idx < RROWS * (K_DIM / 8); idx += NTHREAD) {
    int r  = idx / (K_DIM / 8);
    int k8 = (idx % (K_DIM / 8)) * 8;
    int gr = (r < 3 * MS) ? ((r / MS) * M_DIM + m0 + (r % MS))
                          : (3 * M_DIM + m0 + (r - 3 * MS));
    *(v8bf*)&lW[r][k8] = *(const v8bf*)&Wcomb[(size_t)gr * K_DIM + k8];
  }
  // ---- Init c slice from c0 ----
  for (int idx = tid; idx < B_DIM * MS; idx += NTHREAD) {
    int b = idx / MS, m = idx % MS;
    lC[b][m] = c0[b * M_DIM + m0 + m];
  }
  __syncthreads();

  // Wave -> output tile mapping: 64x32 output = 4 btiles x 2 rtiles.
  const int bt    = (wave >> 1) * 16;  // batch tile origin
  const int rt    = (wave & 1) * 16;   // weight-row tile origin
  const int lhalf = lane >> 4;         // 0/1 lane half
  const int l16   = lane & 15;

  for (int t = 0; t < T_DIM; ++t) {
    // Async-DMA one K-chunk of [x_t | h] into lXH[ch&1] (pure copy, no VGPRs).
    auto issue_chunk = [&](int ch) {
      const int kbase = ch * CHUNK;
      const int buf   = ch & 1;
      const __bf16* src;
      int stride;
      if (kbase < IN_DIM) {
        src    = xbf + (size_t)t * (B_DIM * IN_DIM) + kbase;
        stride = IN_DIM;
      } else {
        src    = hbuf + (kbase - IN_DIM);
        stride = M_DIM;
      }
      for (int seg = tid; seg < B_DIM * (CHUNK / 8); seg += NTHREAD) {
        int b  = seg >> 5;            // / (CHUNK/8)
        int e8 = (seg & 31) * 8;      // bf16 element offset, 16B granules
        uint64_t ga = (uint64_t)(uintptr_t)(src + (size_t)b * stride + e8);
        uint32_t la = (uint32_t)(uintptr_t)&lXH[buf][b][e8];
        asm volatile("global_load_async_to_lds_b128 %0, %1, off"
                     :: "v"(la), "v"(ga)
                     : "memory");
      }
    };

    v8f acc = {};
    issue_chunk(0);
    asm volatile("s_wait_asynccnt 0" ::: "memory");
    __syncthreads();

    for (int ch = 0; ch < NCHUNK; ++ch) {
      const int kbase = ch * CHUNK;
      const int buf   = ch & 1;
      if (ch + 1 < NCHUNK) issue_chunk(ch + 1);  // overlap DMA with WMMA

#pragma unroll
      for (int ks = 0; ks < CHUNK / KSTEP; ++ks) {
        const int k0 = ks * KSTEP;
        // A fragment: 16x32 bf16. lanes 0-15: K 0..15; lanes 16-31: K 16..31
        v8bf alo = *(const v8bf*)&lXH[buf][bt + l16][k0 + lhalf * 16];
        v8bf ahi = *(const v8bf*)&lXH[buf][bt + l16][k0 + lhalf * 16 + 8];
        v16bf a = __builtin_shufflevector(alo, ahi, 0, 1, 2, 3, 4, 5, 6, 7,
                                          8, 9, 10, 11, 12, 13, 14, 15);
        // B fragment: 32x16 bf16; lane n holds 16 contiguous K of its column.
        v16bf bfrag =
            *(const v16bf*)&lW[rt + l16][kbase + k0 + lhalf * 16];
        acc = __builtin_amdgcn_wmma_f32_16x16x32_bf16(
            false, a, false, bfrag, (short)0, acc, false, false);
      }
      asm volatile("s_wait_asynccnt 0" ::: "memory");
      __syncthreads();
    }

    // ---- Scatter D tiles to LDS: lane n -> col, vgpr v -> row (M=lhalf*8+v) ----
#pragma unroll
    for (int v = 0; v < 8; ++v) {
      lG[bt + lhalf * 8 + v][rt + l16] = acc[v];
    }
    __syncthreads();

    // ---- Gate math + state update on this WG's hidden slice ----
    for (int idx = tid; idx < B_DIM * MS; idx += NTHREAD) {
      int b = idx / MS, m = idx % MS;
      int mg = m0 + m;
      float iv = lG[b][0 * MS + m] + biasc[mg];
      float ov = lG[b][1 * MS + m] + biasc[M_DIM + mg];
      float uv = lG[b][2 * MS + m] + biasc[2 * M_DIM + mg];
      float fv = lG[b][3 * MS + m] + biasc[3 * M_DIM + mg];
      iv = 1.0f / (1.0f + __expf(-iv));
      ov = 1.0f / (1.0f + __expf(-ov));
      uv = tanhf(uv);
      fv = 1.0f / (1.0f + __expf(-fv));
      float cn = iv * uv + fv * lC[b][m];
      lC[b][m] = cn;
      float hn = ov * tanhf(cn);
      out[(size_t)t * (B_DIM * M_DIM) + b * M_DIM + mg] = hn;
      lHs[b][m] = (__bf16)hn;
    }
    __syncthreads();

    // ---- Publish h slice: LDS -> global hbuf via async store (16B per row) ----
    if (tid < B_DIM) {
      uint64_t ga = (uint64_t)(uintptr_t)(hbuf + (size_t)tid * M_DIM + m0);
      uint32_t la = (uint32_t)(uintptr_t)&lHs[tid][0];
      asm volatile("global_store_async_from_lds_b128 %0, %1, off"
                   :: "v"(ga), "v"(la)
                   : "memory");
    }
    asm volatile("s_wait_asynccnt 0" ::: "memory");

    // Prefetch next timestep's x into L2 while we wait at the barrier.
    if (t + 1 < T_DIM) {
      const __bf16* xn = xbf + (size_t)(t + 1) * (B_DIM * IN_DIM);
      __builtin_prefetch(xn + tid * 128, 0, 0);
    }

    // ---- Grid-wide step barrier (monotonic counter, device scope) ----
    __threadfence();
    __syncthreads();
    if (tid == 0) {
      __hip_atomic_fetch_add(ctr, 1u, __ATOMIC_ACQ_REL,
                             __HIP_MEMORY_SCOPE_AGENT);
      const unsigned int target = (unsigned int)(t + 1) * NWG;
      while (__hip_atomic_load(ctr, __ATOMIC_ACQUIRE,
                               __HIP_MEMORY_SCOPE_AGENT) < target) {
        __builtin_amdgcn_s_sleep(2);
      }
    }
    __syncthreads();
    __threadfence();
  }

  // ---- Final state outputs: c_final then h_final ----
  float* cfin = out + (size_t)T_DIM * B_DIM * M_DIM;
  float* hfin = cfin + B_DIM * M_DIM;
  for (int idx = tid; idx < B_DIM * MS; idx += NTHREAD) {
    int b = idx / MS, m = idx % MS;
    int mg = m0 + m;
    cfin[b * M_DIM + mg] = lC[b][m];
    hfin[b * M_DIM + mg] =
        out[(size_t)(T_DIM - 1) * (B_DIM * M_DIM) + b * M_DIM + mg];
  }
}

// ---------------------------------------------------------------------------
// Host launcher
// ---------------------------------------------------------------------------
extern "C" void kernel_launch(void* const* d_in, const int* in_sizes, int n_in,
                              void* d_out, int out_size, void* d_ws,
                              size_t ws_size, hipStream_t stream) {
  (void)in_sizes; (void)n_in; (void)out_size; (void)ws_size;

  const float* x      = (const float*)d_in[0];   // [T,B,IN]
  const float* h0     = (const float*)d_in[1];   // [B,M]
  const float* c0     = (const float*)d_in[2];   // [B,M]
  const float* W_ioux = (const float*)d_in[3];   // [3M,IN]
  const float* b_ioux = (const float*)d_in[4];   // [3M]
  const float* W_iouh = (const float*)d_in[5];   // [3M,M]
  const float* b_iouh = (const float*)d_in[6];   // [3M]
  const float* W_fx   = (const float*)d_in[7];   // [M,IN]
  const float* b_fx   = (const float*)d_in[8];   // [M]
  const float* W_fh   = (const float*)d_in[9];   // [M,M]
  const float* b_fh   = (const float*)d_in[10];  // [M]
  float* out = (float*)d_out;

  // Workspace layout (256B-aligned offsets):
  char* ws = (char*)d_ws;
  unsigned int* ctr = (unsigned int*)(ws + 0);
  __bf16* hbuf  = (__bf16*)(ws + 1024);                        // 64 KB
  float*  biasc = (float*)(ws + 1024 + 65536);                 // 8 KB
  __bf16* Wcomb = (__bf16*)(ws + 1024 + 65536 + 8192);         // 4 MB
  __bf16* xbf   = (__bf16*)(ws + 1024 + 65536 + 8192 + 4194304); // 32 MB

  // 1) init counter / h0 / biases
  lstm_init_kernel<<<(B_DIM * M_DIM + 255) / 256, 256, 0, stream>>>(
      h0, b_ioux, b_iouh, b_fx, b_fh, hbuf, biasc, ctr);
  // 2) convert x to bf16 (parallel prepass, ~4us of HBM traffic)
  lstm_convert_x_kernel<<<(T_DIM * B_DIM * IN_DIM / 4 + 255) / 256, 256, 0,
                          stream>>>(x, xbf);
  // 3) pack combined weights to bf16
  lstm_pack_weights_kernel<<<(4 * M_DIM * K_DIM + 255) / 256, 256, 0, stream>>>(
      W_ioux, W_iouh, W_fx, W_fh, Wcomb);
  // 4) persistent fused scan
  lstm_scan_kernel<<<NWG, NTHREAD, 0, stream>>>(xbf, c0, Wcomb, biasc, hbuf,
                                                ctr, out);
}
